// CrossAttentionFusion_31662498906304
// MI455X (gfx1250) — matrix operand order repaired and checked by reading, not compiled, not run on hardware
//
#include <hip/hip_runtime.h>
#include <hip/hip_bf16.h>

// ---------------------------------------------------------------------------
// CrossAttentionFusion on MI455X (gfx1250).
// - All matmuls on v_wmma_f32_16x16x32_bf16 (wave32 WMMA), f32 accumulate.
// - Activations/weights pre-converted to bf16 so every GEMM operand stage is
//   a 16B copy -> eligible for GLOBAL_LOAD_ASYNC_TO_LDS_B128 (ASYNCcnt),
//   double-buffered to overlap HBM/L2 traffic with the WMMA stream.
// - vit_to_graph MHA is dead code in the reference and is skipped.
// ---------------------------------------------------------------------------

typedef __bf16 bf16;
typedef __attribute__((ext_vector_type(16))) __bf16 bf16x16;
typedef __attribute__((ext_vector_type(8)))  __bf16 bf16x8;
typedef __attribute__((ext_vector_type(4)))  __bf16 bf16x4;
typedef __attribute__((ext_vector_type(8)))  float  f32x8;
typedef __attribute__((ext_vector_type(4)))  int    i32x4;

#define DD 1024
#define SS 1024
#define BB 8
#define HH 16
#define HD 64
#define NROWS (BB * SS)   // 8192

// ---- CDNA5 async global->LDS copy (tracked by ASYNCcnt), with fallback ----
#if __has_builtin(__builtin_amdgcn_global_load_async_to_lds_b128)
#define HAVE_ASYNC_LDS 1
#else
#define HAVE_ASYNC_LDS 0
#endif

#if __has_builtin(__builtin_amdgcn_s_wait_asynccnt)
#define WAIT_ASYNC(n) __builtin_amdgcn_s_wait_asynccnt(n)
#elif defined(__AMDGCN__)
#define WAIT_ASYNC(n) asm volatile("s_wait_asynccnt %0" ::"n"(n) : "memory")
#else
#define WAIT_ASYNC(n)
#endif

typedef __attribute__((address_space(1))) i32x4 as1_i32x4;
typedef __attribute__((address_space(3))) i32x4 as3_i32x4;

__device__ __forceinline__ void async_cp16(const bf16* g, bf16* l) {
#if HAVE_ASYNC_LDS
    __builtin_amdgcn_global_load_async_to_lds_b128(
        (as1_i32x4*)(g), (as3_i32x4*)(l), 0, 0);
#else
    *(bf16x8*)l = *(const bf16x8*)g;
#endif
}

__device__ __forceinline__ f32x8 wmma_bf16(bf16x16 a, bf16x16 b, f32x8 c) {
    return __builtin_amdgcn_wmma_f32_16x16x32_bf16(false, a, false, b,
                                                   (short)0, c, false, false);
}

__device__ __forceinline__ bf16x16 cat8(bf16x8 a, bf16x8 b) {
    return __builtin_shufflevector(a, b, 0, 1, 2, 3, 4, 5, 6, 7,
                                   8, 9, 10, 11, 12, 13, 14, 15);
}

// Butterfly reductions confined to 16-lane half-waves (WMMA C layout:
// lanes 0-15 hold rows 0-7, lanes 16-31 hold rows 8-15).
__device__ __forceinline__ float half_max(float v) {
    for (int off = 1; off < 16; off <<= 1) v = fmaxf(v, __shfl_xor(v, off, 32));
    return v;
}
__device__ __forceinline__ float half_sum(float v) {
    for (int off = 1; off < 16; off <<= 1) v += __shfl_xor(v, off, 32);
    return v;
}

// ---------------------------------------------------------------------------
// f32 -> bf16 conversion (weights + activations; one-time, ~6us of traffic)
// ---------------------------------------------------------------------------
__global__ __launch_bounds__(256) void cvt_f32_bf16_kernel(
    const float* __restrict__ in, bf16* __restrict__ out, int n4) {
    int i = (blockIdx.x * 256 + threadIdx.x);
    if (i >= n4) return;
    const float4 v = ((const float4*)in)[i];
    bf16x4 o;
    o[0] = (bf16)v.x; o[1] = (bf16)v.y; o[2] = (bf16)v.z; o[3] = (bf16)v.w;
    ((bf16x4*)out)[i] = o;
}

// ---------------------------------------------------------------------------
// Y[M,N] = scale * ( A[M,K] @ W[N,K]^T + bias[N] )
// A is the virtual bf16 concat [A0 | A1 | A2] (k-split at 1024/2048), each
// source row-stride 1024.  Tiles: BM=BN=128, BK=32; 8 waves in a 2x4 grid,
// each wave owns 4x2 WMMA subtiles (64 f32 accum VGPRs/lane).
// LDS tiles are double-buffered and filled by async global->LDS b128 copies:
// tile k+1 is issued before waiting (s_wait_asynccnt 4) on tile k.
// OUT_MODE: 0 = f32 row-major, 1 = bf16 row-major,
//           2 = bf16 transposed per-head  Vt[b][h][d][s]
// ---------------------------------------------------------------------------
template <int OUT_MODE>
__global__ __launch_bounds__(256) void gemm_bias_kernel(
    const bf16* __restrict__ A0, const bf16* __restrict__ A1,
    const bf16* __restrict__ A2, const bf16* __restrict__ W,
    const float* __restrict__ bias, void* __restrict__ outv,
    float scale, int M, int N, int K) {
    __shared__ __align__(16) bf16 ldsA[2][128][48];
    __shared__ __align__(16) bf16 ldsW[2][128][48];

    const int t = threadIdx.x;
    const int wave = t >> 5, lane = t & 31;
    const int hi = lane >> 4, lo = lane & 15;
    const int wm = wave >> 2, wn = wave & 3;
    const int m0 = blockIdx.y * 128, n0 = blockIdx.x * 128;

    f32x8 acc[4][2] = {};

    const int ar = t >> 1;         // tile row this thread stages (0..127)
    const int ac = (t & 1) * 16;   // k-chunk 0 or 16

    // issue the 4 async copies (2 A + 2 W) for one 128x32 tile pair
    auto stage = [&](int buf, int k0) {
        const int kg = k0 + ac;
        const bf16* Ap = (kg < 1024)   ? A0 + (size_t)(m0 + ar) * DD + kg
                         : (kg < 2048) ? A1 + (size_t)(m0 + ar) * DD + (kg - 1024)
                                       : A2 + (size_t)(m0 + ar) * DD + (kg - 2048);
        async_cp16(Ap,     &ldsA[buf][ar][ac]);
        async_cp16(Ap + 8, &ldsA[buf][ar][ac + 8]);
        const bf16* Wp = W + (size_t)(n0 + ar) * K + k0 + ac;
        async_cp16(Wp,     &ldsW[buf][ar][ac]);
        async_cp16(Wp + 8, &ldsW[buf][ar][ac + 8]);
    };

    const int NK = K >> 5;
    stage(0, 0);
    for (int ks = 0; ks < NK; ++ks) {
        const int buf = ks & 1;
        if (ks + 1 < NK) {
            stage(buf ^ 1, (ks + 1) * 32);
            WAIT_ASYNC(4);   // tile ks complete; tile ks+1 still in flight
        } else {
            WAIT_ASYNC(0);
        }
        __syncthreads();

        // fragments per ISA 7.12.2 layouts
        bf16x16 af[4], bfr[2];
#pragma unroll
        for (int i = 0; i < 4; ++i) {
            const bf16* p = &ldsA[buf][wm * 64 + i * 16 + lo][hi * 8];
            af[i] = cat8(*(const bf16x8*)p, *(const bf16x8*)(p + 16));
        }
#pragma unroll
        for (int j = 0; j < 2; ++j) {
            const bf16* p = &ldsW[buf][wn * 32 + j * 16 + lo][hi * 8];
            bfr[j] = cat8(*(const bf16x8*)p, *(const bf16x8*)(p + 16));
        }
#pragma unroll
        for (int i = 0; i < 4; ++i)
#pragma unroll
            for (int j = 0; j < 2; ++j)
                acc[i][j] = wmma_bf16(af[i], bfr[j], acc[i][j]);
        __syncthreads();   // reads done before next iter overwrites buf^1
    }

    // ---- epilogue: (acc + bias) * scale, store per mode
#pragma unroll
    for (int j = 0; j < 2; ++j) {
        const int col = n0 + wn * 32 + j * 16 + lo;
        const float bv = bias[col];
#pragma unroll
        for (int i = 0; i < 4; ++i) {
            const int rbase = m0 + wm * 64 + i * 16 + hi * 8;
            if (OUT_MODE == 0) {
                float* o = (float*)outv;
#pragma unroll
                for (int e = 0; e < 8; ++e)
                    o[(size_t)(rbase + e) * N + col] = (acc[i][j][e] + bv) * scale;
            } else if (OUT_MODE == 1) {
                bf16* o = (bf16*)outv;
#pragma unroll
                for (int e = 0; e < 8; ++e)
                    o[(size_t)(rbase + e) * N + col] =
                        (bf16)((acc[i][j][e] + bv) * scale);
            } else {
                // Vt[b][h][d][s]; 8 consecutive s values -> one b128 store
                bf16* o = (bf16*)outv;
                const int h = col >> 6, d = col & 63;
                const int bb = rbase >> 10, s = rbase & 1023;
                bf16x8 v;
#pragma unroll
                for (int e = 0; e < 8; ++e)
                    v[e] = (bf16)((acc[i][j][e] + bv) * scale);
                *(bf16x8*)(o + (((size_t)bb * HH + h) * HD + d) * SS + s) = v;
            }
        }
    }
}

// ---------------------------------------------------------------------------
// Flash attention, one (b,h) per blockIdx.y, 128 query rows per workgroup,
// 16 query rows per wave (row stats stay in a 16-lane half-wave).
// Q,K bf16 row-major [token, D] (Q pre-scaled by 1/8); V^T bf16 [b][h][d][s].
// K/V tiles double-buffered via async global->LDS copies.
// ---------------------------------------------------------------------------
__global__ __launch_bounds__(256) void flash_attn_kernel(
    const bf16* __restrict__ Q, const bf16* __restrict__ Kb,
    const bf16* __restrict__ Vt, bf16* __restrict__ O) {
    __shared__ __align__(16) bf16 ldsK[2][32][72];
    __shared__ __align__(16) bf16 ldsV[2][64][48];
    __shared__ __align__(16) bf16 ldsP[8][16][48];

    const int t = threadIdx.x, wave = t >> 5, lane = t & 31;
    const int hi = lane >> 4, lo = lane & 15;
    const int bh = blockIdx.y, b = bh >> 4, h = bh & 15;
    const int q0 = blockIdx.x * 128 + wave * 16;

    // Persistent Q A-fragments (contraction dim hd=64 -> 2 frags of K=32)
    bf16x16 aQ[2];
    {
        const size_t qoff = (size_t)(b * SS + q0 + lo) * DD + h * HD;
#pragma unroll
        for (int kk = 0; kk < 2; ++kk) {
            const bf16* p = Q + qoff + kk * 32 + hi * 8;
            aQ[kk] = cat8(*(const bf16x8*)p, *(const bf16x8*)(p + 16));
        }
    }

    float mrow[8], lrow[8];
#pragma unroll
    for (int e = 0; e < 8; ++e) { mrow[e] = -1e30f; lrow[e] = 0.f; }
    f32x8 accO[4] = {};

    const int kr = t >> 3, kc = (t & 7) * 8;  // K stage: 32 x 64
    const int vr = t >> 2, vc = (t & 3) * 8;  // Vt stage: 64 x 32

    auto stageKV = [&](int buf, int kt) {
        async_cp16(Kb + (size_t)(b * SS + kt + kr) * DD + h * HD + kc,
                   &ldsK[buf][kr][kc]);
        async_cp16(Vt + ((size_t)(b * HH + h) * HD + vr) * SS + kt + vc,
                   &ldsV[buf][vr][vc]);
    };

    stageKV(0, 0);
    for (int kt = 0; kt < SS; kt += 32) {
        const int buf = (kt >> 5) & 1;
        if (kt + 32 < SS) {
            stageKV(buf ^ 1, kt + 32);
            WAIT_ASYNC(2);
        } else {
            WAIT_ASYNC(0);
        }
        __syncthreads();

        // scores: S = Q @ K^T  (two 16x16 col tiles, hd=64 -> 2 wmma steps)
        f32x8 s[2] = {};
#pragma unroll
        for (int j = 0; j < 2; ++j)
#pragma unroll
            for (int kk = 0; kk < 2; ++kk) {
                const bf16* p = &ldsK[buf][j * 16 + lo][kk * 32 + hi * 8];
                bf16x16 bK = cat8(*(const bf16x8*)p, *(const bf16x8*)(p + 16));
                s[j] = wmma_bf16(aQ[kk], bK, s[j]);
            }

        // online softmax per query row; write P (bf16) into per-wave LDS
#pragma unroll
        for (int e = 0; e < 8; ++e) {
            const float v0 = s[0][e], v1 = s[1][e];   // scale folded into Q
            const float mx = half_max(fmaxf(v0, v1));
            const float mn = fmaxf(mrow[e], mx);
            const float fs = __expf(mrow[e] - mn);
            const float p0 = __expf(v0 - mn), p1 = __expf(v1 - mn);
            const float ls = half_sum(p0 + p1);
            lrow[e] = lrow[e] * fs + ls;
            mrow[e] = mn;
#pragma unroll
            for (int nt = 0; nt < 4; ++nt) accO[nt][e] *= fs;
            ldsP[wave][e + hi * 8][lo]      = (bf16)p0;
            ldsP[wave][e + hi * 8][16 + lo] = (bf16)p1;
        }
        __syncthreads();

        // out += P @ V   (contraction K=32 -> one wmma per 16-col d subtile)
        const bf16* pp = &ldsP[wave][lo][hi * 8];
        const bf16x16 aP = cat8(*(const bf16x8*)pp, *(const bf16x8*)(pp + 16));
#pragma unroll
        for (int nt = 0; nt < 4; ++nt) {
            const bf16* vp = &ldsV[buf][nt * 16 + lo][hi * 8];
            bf16x16 bV = cat8(*(const bf16x8*)vp, *(const bf16x8*)(vp + 16));
            accO[nt] = wmma_bf16(aP, bV, accO[nt]);
        }
        __syncthreads();   // reads done before next iter overwrites buf^1
    }

    // normalize and store O (bf16 row-major [token, D])
#pragma unroll
    for (int e = 0; e < 8; ++e) {
        const float inv = 1.f / lrow[e];
        const size_t r = (size_t)(b * SS + q0 + e + hi * 8);
#pragma unroll
        for (int nt = 0; nt < 4; ++nt)
            O[r * DD + h * HD + nt * 16 + lo] = (bf16)(accO[nt][e] * inv);
    }
}

// ---------------------------------------------------------------------------
// In-place LayerNorm + ReLU over rows of 1024 f32 (x fully rewritten from
// inputs each call, so in-place is graph-replay safe).
// ---------------------------------------------------------------------------
__global__ __launch_bounds__(256) void ln_relu_kernel(
    float* __restrict__ x, const float* __restrict__ g,
    const float* __restrict__ beta) {
    __shared__ float redS[8], redS2[8];
    const int t = threadIdx.x;
    const size_t base = (size_t)blockIdx.x * DD + t * 4;
    const float4 v = *(const float4*)(x + base);
    float s = v.x + v.y + v.z + v.w;
    float s2 = v.x * v.x + v.y * v.y + v.z * v.z + v.w * v.w;
    for (int off = 1; off < 32; off <<= 1) {
        s += __shfl_xor(s, off, 32);
        s2 += __shfl_xor(s2, off, 32);
    }
    if ((t & 31) == 0) { redS[t >> 5] = s; redS2[t >> 5] = s2; }
    __syncthreads();
    float tot = 0.f, tot2 = 0.f;
#pragma unroll
    for (int i = 0; i < 8; ++i) { tot += redS[i]; tot2 += redS2[i]; }
    const float mu = tot * (1.f / 1024.f);
    const float var = tot2 * (1.f / 1024.f) - mu * mu;
    const float rs = rsqrtf(var + 1e-5f);
    const int c = t * 4;
    float4 o;
    o.x = fmaxf((v.x - mu) * rs * g[c + 0] + beta[c + 0], 0.f);
    o.y = fmaxf((v.y - mu) * rs * g[c + 1] + beta[c + 1], 0.f);
    o.z = fmaxf((v.z - mu) * rs * g[c + 2] + beta[c + 2], 0.f);
    o.w = fmaxf((v.w - mu) * rs * g[c + 3] + beta[c + 3], 0.f);
    *(float4*)(x + base) = o;
}

// ---------------------------------------------------------------------------
// Launcher.
// Input layout (setup_inputs() insertion order, leaves in dict order):
//  0 vit_features  1 graph_features  2 anatomical_features
//  3..10  vit_to_graph {q,k,v,o} x {w,b}   (UNUSED: result is dead code)
//  11..18 graph_to_vit {q,k,v,o} x {w,b}
//  19 fusion.lin.w [1024,3072]  20 fusion.lin.b  21 ln_g  22 ln_b
// ---------------------------------------------------------------------------
extern "C" void kernel_launch(void* const* d_in, const int* in_sizes, int n_in,
                              void* d_out, int out_size, void* d_ws,
                              size_t ws_size, hipStream_t stream) {
    (void)in_sizes; (void)n_in; (void)out_size; (void)ws_size;

    const float* vit   = (const float*)d_in[0];
    const float* graph = (const float*)d_in[1];
    const float* anat  = (const float*)d_in[2];
    const float* Wq = (const float*)d_in[11]; const float* bq = (const float*)d_in[12];
    const float* Wk = (const float*)d_in[13]; const float* bk = (const float*)d_in[14];
    const float* Wv = (const float*)d_in[15]; const float* bv = (const float*)d_in[16];
    const float* Wo = (const float*)d_in[17]; const float* bo = (const float*)d_in[18];
    const float* Wf = (const float*)d_in[19]; const float* bf = (const float*)d_in[20];
    const float* lng = (const float*)d_in[21]; const float* lnb = (const float*)d_in[22];

    constexpr size_t MB = 1024u * 1024u;
    char* ws = (char*)d_ws;
    bf16* WQb   = (bf16*)(ws + 0 * MB);    // 2MB each (1024x1024 bf16)
    bf16* WKb   = (bf16*)(ws + 2 * MB);
    bf16* WVb   = (bf16*)(ws + 4 * MB);
    bf16* WOb   = (bf16*)(ws + 6 * MB);
    bf16* WFb   = (bf16*)(ws + 8 * MB);    // 6MB (1024x3072)
    bf16* vitb  = (bf16*)(ws + 14 * MB);   // 16MB each, [8192,1024] bf16
    bf16* grb   = (bf16*)(ws + 30 * MB);
    bf16* anb   = (bf16*)(ws + 46 * MB);
    bf16* Qb    = (bf16*)(ws + 62 * MB);
    bf16* Kbf   = (bf16*)(ws + 78 * MB);
    bf16* Vtb   = (bf16*)(ws + 94 * MB);   // transposed [B,H,hd,S]
    bf16* Ob    = (bf16*)(ws + 110 * MB);
    bf16* Projb = (bf16*)(ws + 126 * MB);

    // 1) weights + activations -> bf16
    cvt_f32_bf16_kernel<<<1024, 256, 0, stream>>>(Wq, WQb, (DD * DD) / 4);
    cvt_f32_bf16_kernel<<<1024, 256, 0, stream>>>(Wk, WKb, (DD * DD) / 4);
    cvt_f32_bf16_kernel<<<1024, 256, 0, stream>>>(Wv, WVb, (DD * DD) / 4);
    cvt_f32_bf16_kernel<<<1024, 256, 0, stream>>>(Wo, WOb, (DD * DD) / 4);
    cvt_f32_bf16_kernel<<<3072, 256, 0, stream>>>(Wf, WFb, (3 * DD * DD) / 4);
    cvt_f32_bf16_kernel<<<8192, 256, 0, stream>>>(vit, vitb, (NROWS * DD) / 4);
    cvt_f32_bf16_kernel<<<8192, 256, 0, stream>>>(graph, grb, (NROWS * DD) / 4);
    cvt_f32_bf16_kernel<<<8192, 256, 0, stream>>>(anat, anb, (NROWS * DD) / 4);

    // 2) QKV projections (graph_to_vit: Q from graph, K/V from vit).
    //    Attention scale 1/sqrt(64) folded into the Q epilogue.
    const dim3 gg(DD / 128, NROWS / 128);  // (8, 64)
    gemm_bias_kernel<1><<<gg, 256, 0, stream>>>(
        grb, nullptr, nullptr, WQb, bq, Qb, 0.125f, NROWS, DD, DD);
    gemm_bias_kernel<1><<<gg, 256, 0, stream>>>(
        vitb, nullptr, nullptr, WKb, bk, Kbf, 1.0f, NROWS, DD, DD);
    gemm_bias_kernel<2><<<gg, 256, 0, stream>>>(
        vitb, nullptr, nullptr, WVb, bv, Vtb, 1.0f, NROWS, DD, DD);

    // 3) attention
    flash_attn_kernel<<<dim3(SS / 128, BB * HH), 256, 0, stream>>>(Qb, Kbf, Vtb, Ob);

    // 4) output projection -> bf16 (feeds fusion concat)
    gemm_bias_kernel<1><<<gg, 256, 0, stream>>>(
        Ob, nullptr, nullptr, WOb, bo, Projb, 1.0f, NROWS, DD, DD);

    // 5) fusion GEMM over virtual concat [vit | proj | anat], x -> d_out
    gemm_bias_kernel<0><<<gg, 256, 0, stream>>>(
        vitb, Projb, anb, WFb, bf, d_out, 1.0f, NROWS, DD, 3 * DD);

    // 6) LayerNorm + ReLU in place on d_out
    ln_relu_kernel<<<NROWS, 256, 0, stream>>>((float*)d_out, lng, lnb);
}
